// SpatialAttention_18717467476579
// MI455X (gfx1250) — matrix operand order repaired and checked by reading, not compiled, not run on hardware
//
#include <hip/hip_runtime.h>

typedef __attribute__((ext_vector_type(16))) __bf16 v16bf;
typedef __attribute__((ext_vector_type(8)))  __bf16 v8bf;
typedef __attribute__((ext_vector_type(8)))  float  v8f;

#define BT    128      // b*t
#define NTOK  197      // 1 + 14*14
#define NPAD  224      // 7*32
#define C0    768
#define C3    2304
#define NH    12
#define HD    64
#define MROWS 25216    // BT*NTOK, == 16*1576 == 32*788
#define MT    1576
#define BTH   1536     // BT*NH

#define WMMA_BF16(A,B,C) __builtin_amdgcn_wmma_f32_16x16x32_bf16(false,(A),false,(B),(short)0,(C),false,false)

// A-operand (16x32 bf16): lane l<16 row=l holds K {k0..k0+7, k0+16..k0+23};
// lane l>=16 row=l-16 holds K {k0+8..k0+15, k0+24..k0+31}.
static __device__ inline v16bf load_a16(const __bf16* rowk0, int half) {
  int kb = half ? 8 : 0;
  union { v8bf h[2]; v16bf v; } u;
  u.h[0] = *(const v8bf*)(rowk0 + kb);
  u.h[1] = *(const v8bf*)(rowk0 + kb + 16);
  return u.v;
}
// B-operand (32x16 bf16): lane l<16 col=l holds K {k0..k0+7, k0+16..k0+23}; lane>=16 col holds {k0+8..k0+15, k0+24..k0+31}.
// Source stored "column-major" (one row per output column), so loads are contiguous.
static __device__ inline v16bf load_b16(const __bf16* colk0, int half) {
  int kb = half ? 8 : 0;
  union { v8bf h[2]; v16bf v; } u;
  u.h[0] = *(const v8bf*)(colk0 + kb);
  u.h[1] = *(const v8bf*)(colk0 + kb + 16);
  return u.v;
}

__global__ void k_cvt(const float* __restrict__ in, __bf16* __restrict__ out, long n) {
  long i = (long)blockIdx.x * blockDim.x + threadIdx.x;
  long st = (long)gridDim.x * blockDim.x;
  for (; i < n; i += st) out[i] = (__bf16)in[i];
}

__global__ void k_transpose_cvt(const float* __restrict__ in, __bf16* __restrict__ out,
                                int R, int Ccols) {
  long e = (long)blockIdx.x * blockDim.x + threadIdx.x;
  long n = (long)R * Ccols;
  if (e >= n) return;
  int c = (int)(e % Ccols);
  int r = (int)(e / Ccols);
  out[(long)c * R + r] = (__bf16)in[e];
}

// ---------------- QKV GEMM: [25216x768] @ [768x2304] + bias, scatter to Q/K/Vt (bf16)
// Register-blocked: each wave computes a 32x64 macro-tile (2 m-tiles x 4 n-tiles).
// A 64-wide n-group is exactly one (qkv, head) block, so scatter target is wave-uniform.
__global__ void __launch_bounds__(128) k_gemm_qkv(const __bf16* __restrict__ X,
                                                  const __bf16* __restrict__ WT,
                                                  const float*  __restrict__ bias,
                                                  __bf16* __restrict__ Qb,
                                                  __bf16* __restrict__ Kb,
                                                  __bf16* __restrict__ Vt) {
  const int NG = C3 / 64;                      // 36
  const int MG = MROWS / 32;                   // 788
  int wave = (int)((blockIdx.x * blockDim.x + threadIdx.x) >> 5);
  if (wave >= MG * NG) return;
  int lane = threadIdx.x & 31;
  int half = lane >> 4, l15 = lane & 15;
  int mg = wave / NG, ng = wave % NG;
  int m0 = mg * 32, n0 = ng * 64;
  const __bf16* ap0 = X + (size_t)(m0 + l15) * C0;
  const __bf16* ap1 = X + (size_t)(m0 + 16 + l15) * C0;
  const __bf16* bp  = WT + (size_t)(n0 + l15) * C0;
  v8f acc[2][4] = {};
  for (int k0 = 0; k0 < C0; k0 += 32) {
    v16bf a0 = load_a16(ap0 + k0, half);
    v16bf a1 = load_a16(ap1 + k0, half);
#pragma unroll
    for (int nt = 0; nt < 4; ++nt) {
      v16bf b = load_b16(bp + (size_t)nt * 16 * C0 + k0, half);
      acc[0][nt] = WMMA_BF16(a0, b, acc[0][nt]);
      acc[1][nt] = WMMA_BF16(a1, b, acc[1][nt]);
    }
  }
  int qkvi = n0 / C0;                          // wave-uniform
  int head = (n0 % C0) / HD;                   // wave-uniform
#pragma unroll
  for (int nt = 0; nt < 4; ++nt) {
    int dd = nt * 16 + l15;
    float bn = bias[n0 + dd];
#pragma unroll
    for (int mi = 0; mi < 2; ++mi) {
#pragma unroll
      for (int r = 0; r < 8; ++r) {
        int m = m0 + mi * 16 + r + 8 * half;
        int bt = m / NTOK;
        int tok = m - bt * NTOK;
        int bth = bt * NH + head;
        __bf16 bv = (__bf16)(acc[mi][nt][r] + bn);
        if (qkvi == 0)      Qb[((size_t)bth * NPAD + tok) * HD + dd] = bv;
        else if (qkvi == 1) Kb[((size_t)bth * NPAD + tok) * HD + dd] = bv;
        else                Vt[((size_t)bth * HD + dd) * NPAD + tok] = bv;
      }
    }
  }
}

// ---------------- rel_h / rel_w: (bth, qs, k) = dot(q[bth,1+qs,:], rel_pos[dist,:])
__global__ void k_rel(const __bf16* __restrict__ Qb,
                      const float* __restrict__ rph, const float* __restrict__ rpw,
                      float* __restrict__ relh, float* __restrict__ relw) {
  int which = blockIdx.y;
  long e = (long)blockIdx.x * blockDim.x + threadIdx.x;
  const long total = (long)BTH * 196 * 14;
  if (e >= total) return;
  int kx  = (int)(e % 14);
  int qs  = (int)((e / 14) % 196);
  int bth = (int)(e / (14 * 196));
  int x   = which ? (qs % 14) : (qs / 14);
  const float* row = (which ? rpw : rph) + (x - kx + 13) * HD;
  const __bf16* q = Qb + ((size_t)bth * NPAD + 1 + qs) * HD;
  float s = 0.f;
#pragma unroll
  for (int d = 0; d < HD; ++d) s += (float)q[d] * row[d];
  (which ? relw : relh)[e] = s;
}

// ---------------- attention: one wave per (bth, qtile)
__global__ void __launch_bounds__(32) k_attn(const __bf16* __restrict__ Qb,
                                             const __bf16* __restrict__ Kb,
                                             const __bf16* __restrict__ Vt,
                                             const float* __restrict__ relh,
                                             const float* __restrict__ relw,
                                             __bf16* __restrict__ Obuf) {
  __shared__ __bf16 Pl[16 * NPAD];
  int qtile = blockIdx.x;          // 0..12
  int bth   = blockIdx.y;          // 0..1535
  int lane = threadIdx.x;
  int half = lane >> 4, l15 = lane & 15;
  const float scale = 0.125f;      // 64^-0.5

  const __bf16* qrow = Qb + ((size_t)bth * NPAD + qtile * 16 + l15) * HD;
  v16bf aQ0 = load_a16(qrow, half);
  v16bf aQ1 = load_a16(qrow + 32, half);

  v8f S[13];
#pragma unroll
  for (int t = 0; t < 13; ++t) {
    const __bf16* kcol = Kb + ((size_t)bth * NPAD + t * 16 + l15) * HD;
    v16bf b0 = load_b16(kcol, half);
    v16bf b1 = load_b16(kcol + 32, half);
    v8f acc = {};
    acc = WMMA_BF16(aQ0, b0, acc);
    acc = WMMA_BF16(aQ1, b1, acc);
    int ktok = t * 16 + l15;
    bool kvalid = (ktok < NTOK);
    int ks = ktok - 1;
#pragma unroll
    for (int r = 0; r < 8; ++r) {
      float s;
      if (!kvalid) {
        s = -__builtin_inff();
      } else {
        s = acc[r] * scale;
        int qtok = qtile * 16 + r + 8 * half;
        if (qtok >= 1 && qtok < NTOK && ktok >= 1) {
          int qs = qtok - 1;
          s += relh[((size_t)bth * 196 + qs) * 14 + ks / 14]
             + relw[((size_t)bth * 196 + qs) * 14 + ks % 14];
        }
      }
      acc[r] = s;
    }
    S[t] = acc;
  }

  // row softmax (rows live across the 16 lanes of each half-wave)
  float rinv[8];
#pragma unroll
  for (int r = 0; r < 8; ++r) {
    float m = -__builtin_inff();
#pragma unroll
    for (int t = 0; t < 13; ++t) m = fmaxf(m, S[t][r]);
#pragma unroll
    for (int off = 1; off < 16; off <<= 1) m = fmaxf(m, __shfl_xor(m, off, 32));
    float sum = 0.f;
#pragma unroll
    for (int t = 0; t < 13; ++t) { float p = __expf(S[t][r] - m); S[t][r] = p; sum += p; }
#pragma unroll
    for (int off = 1; off < 16; off <<= 1) sum += __shfl_xor(sum, off, 32);
    rinv[r] = 1.f / sum;
  }

  // write normalized P to LDS (16 x 224, bf16)
#pragma unroll
  for (int t = 0; t < 13; ++t)
#pragma unroll
    for (int r = 0; r < 8; ++r)
      Pl[(r + 8 * half) * NPAD + t * 16 + l15] = (__bf16)(S[t][r] * rinv[r]);
  // zero pad columns 208..223 (rows 0..15)
#pragma unroll
  for (int j = 0; j < 8; ++j)
    Pl[l15 * NPAD + 208 + half * 8 + j] = (__bf16)0.0f;
  __syncthreads();

  // O = P @ V : K-dim = 224 tokens (7 x 32), N-dim = 64 (4 x 16)
  v8f O[4] = {};
#pragma unroll
  for (int kstep = 0; kstep < 7; ++kstep) {
    int k0 = kstep * 32;
    v16bf a = load_a16((const __bf16*)&Pl[l15 * NPAD + k0], half);
#pragma unroll
    for (int nt = 0; nt < 4; ++nt) {
      const __bf16* vcol = Vt + ((size_t)bth * HD + nt * 16 + l15) * NPAD + k0;
      v16bf b = load_b16(vcol, half);
      O[nt] = WMMA_BF16(a, b, O[nt]);
    }
  }

  int bt = bth / NH, head = bth % NH;
#pragma unroll
  for (int nt = 0; nt < 4; ++nt) {
    int c = head * HD + nt * 16 + l15;
#pragma unroll
    for (int r = 0; r < 8; ++r) {
      int qtok = qtile * 16 + r + 8 * half;
      if (qtok < NTOK)
        Obuf[((size_t)(bt * NTOK + qtok)) * C0 + c] = (__bf16)O[nt][r];
    }
  }
}

// ---------------- projection GEMM: [25216x768] @ [768x768] + bias -> fp32 out
// Register-blocked 32x64 macro-tile per wave.
__global__ void __launch_bounds__(128) k_gemm_proj(const __bf16* __restrict__ A,
                                                   const __bf16* __restrict__ WT,
                                                   const float*  __restrict__ bias,
                                                   float* __restrict__ out) {
  const int NG = C0 / 64;                      // 12
  const int MG = MROWS / 32;                   // 788
  int wave = (int)((blockIdx.x * blockDim.x + threadIdx.x) >> 5);
  if (wave >= MG * NG) return;
  int lane = threadIdx.x & 31;
  int half = lane >> 4, l15 = lane & 15;
  int mg = wave / NG, ng = wave % NG;
  int m0 = mg * 32, n0 = ng * 64;
  const __bf16* ap0 = A + (size_t)(m0 + l15) * C0;
  const __bf16* ap1 = A + (size_t)(m0 + 16 + l15) * C0;
  const __bf16* bp  = WT + (size_t)(n0 + l15) * C0;
  v8f acc[2][4] = {};
  for (int k0 = 0; k0 < C0; k0 += 32) {
    v16bf a0 = load_a16(ap0 + k0, half);
    v16bf a1 = load_a16(ap1 + k0, half);
#pragma unroll
    for (int nt = 0; nt < 4; ++nt) {
      v16bf b = load_b16(bp + (size_t)nt * 16 * C0 + k0, half);
      acc[0][nt] = WMMA_BF16(a0, b, acc[0][nt]);
      acc[1][nt] = WMMA_BF16(a1, b, acc[1][nt]);
    }
  }
#pragma unroll
  for (int nt = 0; nt < 4; ++nt) {
    int n = n0 + nt * 16 + l15;
    float bn = bias[n];
#pragma unroll
    for (int mi = 0; mi < 2; ++mi) {
#pragma unroll
      for (int r = 0; r < 8; ++r) {
        int m = m0 + mi * 16 + r + 8 * half;
        out[(size_t)m * C0 + n] = acc[mi][nt][r] + bn;
      }
    }
  }
}

extern "C" void kernel_launch(void* const* d_in, const int* in_sizes, int n_in,
                              void* d_out, int out_size, void* d_ws, size_t ws_size,
                              hipStream_t stream) {
  const float* x      = (const float*)d_in[0];
  // d_in[1], d_in[2] are h, w (always 14)
  const float* w_qkv  = (const float*)d_in[3];
  const float* b_qkv  = (const float*)d_in[4];
  const float* w_proj = (const float*)d_in[5];
  const float* b_proj = (const float*)d_in[6];
  const float* rph    = (const float*)d_in[7];
  const float* rpw    = (const float*)d_in[8];

  char* ws = (char*)d_ws;
  size_t off = 0;
  auto alloc = [&](size_t bytes) {
    char* p = ws + off;
    off = (off + bytes + 255) & ~(size_t)255;
    return p;
  };
  __bf16* Xb     = (__bf16*)alloc((size_t)MROWS * C0 * 2);
  __bf16* WqkvT  = (__bf16*)alloc((size_t)C3 * C0 * 2);
  __bf16* WprojT = (__bf16*)alloc((size_t)C0 * C0 * 2);
  __bf16* Qb     = (__bf16*)alloc((size_t)BTH * NPAD * HD * 2);
  __bf16* Kb     = (__bf16*)alloc((size_t)BTH * NPAD * HD * 2);
  __bf16* Vt     = (__bf16*)alloc((size_t)BTH * HD * NPAD * 2);
  float*  relh   = (float*)alloc((size_t)BTH * 196 * 14 * 4);
  float*  relw   = (float*)alloc((size_t)BTH * 196 * 14 * 4);
  __bf16* Obuf   = (__bf16*)alloc((size_t)MROWS * C0 * 2);

  // zero the token-padded buffers so pad rows/cols contribute nothing
  hipMemsetAsync(Qb, 0, (size_t)BTH * NPAD * HD * 2, stream);
  hipMemsetAsync(Kb, 0, (size_t)BTH * NPAD * HD * 2, stream);
  hipMemsetAsync(Vt, 0, (size_t)BTH * HD * NPAD * 2, stream);

  // precision conversion / weight transpose
  k_cvt<<<2048, 256, 0, stream>>>(x, Xb, (long)MROWS * C0);
  k_transpose_cvt<<<((long)C0 * C3 + 255) / 256, 256, 0, stream>>>(w_qkv, WqkvT, C0, C3);
  k_transpose_cvt<<<((long)C0 * C0 + 255) / 256, 256, 0, stream>>>(w_proj, WprojT, C0, C0);

  // qkv projection (32x64 macro-tiles)
  {
    int waves = (MROWS / 32) * (C3 / 64);      // 28368
    k_gemm_qkv<<<(waves + 3) / 4, 128, 0, stream>>>(Xb, WqkvT, b_qkv, Qb, Kb, Vt);
  }
  // decomposed rel-pos terms
  {
    long tot = (long)BTH * 196 * 14;
    dim3 g((unsigned)((tot + 255) / 256), 2);
    k_rel<<<g, 256, 0, stream>>>(Qb, rph, rpw, relh, relw);
  }
  // fused attention (scores + bias + softmax + @V)
  {
    dim3 g(13, BTH);
    k_attn<<<g, 32, 0, stream>>>(Qb, Kb, Vt, relh, relw, Obuf);
  }
  // output projection (32x64 macro-tiles)
  {
    int waves = (MROWS / 32) * (C0 / 64);      // 9456
    k_gemm_proj<<<(waves + 3) / 4, 128, 0, stream>>>(Obuf, WprojT, b_proj, (float*)d_out);
  }
}
